// ScaledDotAttention_40166534152829
// MI455X (gfx1250) — compile-verified, matrix operand order checked by reading, not run on hardware
//
#include <hip/hip_runtime.h>

#define BATCH 16
#define SEQ   2048
#define DKDIM 64
#define DVDIM 64
#define DMDIM 512

typedef __attribute__((ext_vector_type(16))) __bf16 v16bf;
typedef __attribute__((ext_vector_type(8)))  float  v8f;
typedef __attribute__((ext_vector_type(4)))  int    v4i;

#define AS1 __attribute__((address_space(1)))
#define AS3 __attribute__((address_space(3)))

// ---------------------------------------------------------------------------
// CDNA5 async global->LDS copies (ASYNCcnt), guarded for toolchain support.
// Probe-discovered signatures: b128 takes (v4i AS1*, v4i AS3*, imm, imm),
// b32 takes (int AS1*, int AS3*, imm, imm).
// ---------------------------------------------------------------------------
#if defined(__has_builtin)
#  if __has_builtin(__builtin_amdgcn_global_load_async_to_lds_b128) && \
      __has_builtin(__builtin_amdgcn_global_load_async_to_lds_b32)
#    define HAVE_ASYNC_LDS 1
#  endif
#endif
#ifndef HAVE_ASYNC_LDS
#  define HAVE_ASYNC_LDS 0
#endif

__device__ __forceinline__ void async_cp_b128(const float* g, float* l) {
#if HAVE_ASYNC_LDS
    __builtin_amdgcn_global_load_async_to_lds_b128(
        (AS1 v4i*)g, (AS3 v4i*)l, 0, 0);
#else
    *(float4*)l = *(const float4*)g;
#endif
}

__device__ __forceinline__ void async_cp_b32(const float* g, float* l) {
#if HAVE_ASYNC_LDS
    __builtin_amdgcn_global_load_async_to_lds_b32(
        (AS1 int*)g, (AS3 int*)l, 0, 0);
#else
    *l = *g;
#endif
}

__device__ __forceinline__ void wait_async_all() {
#if HAVE_ASYNC_LDS
#  if __has_builtin(__builtin_amdgcn_s_wait_asynccnt)
    __builtin_amdgcn_s_wait_asynccnt(0);
#  else
    asm volatile("s_wait_asynccnt 0" ::: "memory");
#  endif
#endif
}

// ISA 7.12.2: 16-bit A-matrix 16x32 — element (vgpr j, half p) of lane (h*16+m)
// holds K index 2*(j%4) + 16*(j/4) + 8*h + p.  B uses the same K mapping with
// lane = column.
__device__ __forceinline__ int kmap(int j, int h) {
    return ((j & 3) << 1) + ((j >> 2) << 4) + (h << 3);
}

// Build a bf16 fragment from an f32 LDS row (contiguous along the K mapping).
__device__ __forceinline__ v16bf make_frag(const float* row, int h, int koff) {
    v16bf f;
    #pragma unroll
    for (int j = 0; j < 8; ++j) {
        int kk = kmap(j, h) + koff;
        f[2*j+0] = (__bf16)row[kk + 0];
        f[2*j+1] = (__bf16)row[kk + 1];
    }
    return f;
}

// Reductions across the 16 lanes sharing one row set of a C/D fragment.
__device__ __forceinline__ float redmax16(float v) {
    v = fmaxf(v, __shfl_xor(v, 8, 32));
    v = fmaxf(v, __shfl_xor(v, 4, 32));
    v = fmaxf(v, __shfl_xor(v, 2, 32));
    v = fmaxf(v, __shfl_xor(v, 1, 32));
    return v;
}
__device__ __forceinline__ float redsum16(float v) {
    v += __shfl_xor(v, 8, 32);
    v += __shfl_xor(v, 4, 32);
    v += __shfl_xor(v, 2, 32);
    v += __shfl_xor(v, 1, 32);
    return v;
}

// Stage one 32-key block: K row-major f32 (B128 async), V transposed f32
// (B32 async — per-lane LDS address performs the transpose).
__device__ __forceinline__ void stage_tile(const float* kb, const float* vb,
                                           int kblk, int tid,
                                           float (*K)[72], float (*Vt)[34]) {
    #pragma unroll
    for (int i = 0; i < 4; ++i) {            // 32x64 f32 = 512 x 16B
        int idx = tid + i * 128;
        int key = idx >> 4;
        int d0  = (idx & 15) << 2;
        async_cp_b128(kb + (size_t)(kblk + key) * DKDIM + d0, &K[key][d0]);
    }
    #pragma unroll
    for (int i = 0; i < 16; ++i) {           // 2048 x 4B, transposed scatter
        int idx = tid + i * 128;
        int key = idx >> 6;
        int d   = idx & 63;
        async_cp_b32(vb + (size_t)(kblk + key) * DVDIM + d, &Vt[d][key]);
    }
}

// ---------------------------------------------------------------------------
// Kernel 1: causal flash attention -> context [B, T, DV] (f32, workspace).
// 128 threads = 4 waves; each wave owns 16 query rows of a 64-row tile.
// Double-buffered async K/V staging overlaps DMA with WMMA + softmax.
// ---------------------------------------------------------------------------
__global__ __launch_bounds__(128) void flash_attn_kernel(
    const float* __restrict__ q, const float* __restrict__ kg,
    const float* __restrict__ vg, float* __restrict__ ctx)
{
    __shared__ float  ldsK[2][32][72];   // 18.0 KB  K block f32 [key][d]
    __shared__ float  ldsVt[2][64][34];  // 17.0 KB  V block f32 transposed [d][key]
    __shared__ __bf16 ldsP[4][16][36];   //  4.5 KB  per-wave P re-layout

    const int b     = blockIdx.y;
    const int qt    = blockIdx.x;
    const int qbase = qt * 64;
    const int tid   = threadIdx.x;
    const int wave  = tid >> 5;
    const int lane  = tid & 31;
    const int h     = lane >> 4;
    const int ln    = lane & 15;
    const int qrow0 = qbase + wave * 16;

    const float* qb = q  + ((size_t)b * SEQ) * DKDIM;
    const float* kb = kg + ((size_t)b * SEQ) * DKDIM;
    const float* vb = vg + ((size_t)b * SEQ) * DVDIM;

    // Q A-fragments (K=0..31, 32..63), pre-scaled by 1/sqrt(DK) = 0.125
    v16bf qa0, qa1;
    {
        const float* qr = qb + (size_t)(qrow0 + ln) * DKDIM;
        #pragma unroll
        for (int j = 0; j < 8; ++j) {
            int kk = kmap(j, h);
            qa0[2*j+0] = (__bf16)(qr[kk +  0] * 0.125f);
            qa0[2*j+1] = (__bf16)(qr[kk +  1] * 0.125f);
            qa1[2*j+0] = (__bf16)(qr[kk + 32] * 0.125f);
            qa1[2*j+1] = (__bf16)(qr[kk + 33] * 0.125f);
        }
    }

    float mrow[8], lrow[8];
    v8f o0 = {}, o1 = {}, o2 = {}, o3 = {};
    #pragma unroll
    for (int r = 0; r < 8; ++r) { mrow[r] = -1e30f; lrow[r] = 0.0f; }

    const int nblk = (qbase + 64) >> 5;          // causal: keys <= q-tile end
    stage_tile(kb, vb, 0, tid, ldsK[0], ldsVt[0]);

    for (int it = 0; it < nblk; ++it) {
        const int kblk = it << 5;
        const int cur  = it & 1;
        wait_async_all();          // my stage of buf[cur] done
        __syncthreads();           // everyone's stage done; prior reads retired
        if (it + 1 < nblk)         // overlap next block's DMA with compute
            stage_tile(kb, vb, kblk + 32, tid, ldsK[cur ^ 1], ldsVt[cur ^ 1]);

        const float (*K)[72]  = ldsK[cur];
        const float (*Vt)[34] = ldsVt[cur];

        // S = (Q*scale) K^T for two 16-key column tiles (contract d = 64)
        v8f s0 = {}, s1 = {};
        #pragma unroll
        for (int nt = 0; nt < 2; ++nt) {
            const float* krow = K[nt * 16 + ln];
            v16bf kf0 = make_frag(krow, h, 0);
            v16bf kf1 = make_frag(krow, h, 32);
            v8f acc = {};
            acc = __builtin_amdgcn_wmma_f32_16x16x32_bf16(false, qa0, false, kf0,
                                                          (short)0, acc, false, false);
            acc = __builtin_amdgcn_wmma_f32_16x16x32_bf16(false, qa1, false, kf1,
                                                          (short)0, acc, false, false);
            if (nt == 0) s0 = acc; else s1 = acc;
        }

        // Causal mask (finite sentinel avoids inf-inf NaN in softmax)
        if (kblk + 31 > qrow0) {               // wave-uniform branch
            #pragma unroll
            for (int r = 0; r < 8; ++r) {
                int qr = qrow0 + r + 8 * h;
                if (kblk + ln      > qr) s0[r] = -1e30f;
                if (kblk + 16 + ln > qr) s1[r] = -1e30f;
            }
        }

        // Online softmax update
        float p0[8], p1[8];
        #pragma unroll
        for (int r = 0; r < 8; ++r) {
            float mb    = redmax16(fmaxf(s0[r], s1[r]));
            float mn    = fmaxf(mrow[r], mb);
            float alpha = __expf(mrow[r] - mn);
            mrow[r] = mn;
            p0[r] = __expf(s0[r] - mn);
            p1[r] = __expf(s1[r] - mn);
            lrow[r] = lrow[r] * alpha + redsum16(p0[r] + p1[r]);
            o0[r] *= alpha; o1[r] *= alpha; o2[r] *= alpha; o3[r] *= alpha;
        }

        // Re-layout P through wave-private LDS (C/D layout -> A layout)
        #pragma unroll
        for (int r = 0; r < 8; ++r) {
            ldsP[wave][r + 8*h][ln]      = (__bf16)p0[r];
            ldsP[wave][r + 8*h][ln + 16] = (__bf16)p1[r];
        }
        asm volatile("s_wait_dscnt 0" ::: "memory");   // LDS in-order per wave

        v16bf pa;
        #pragma unroll
        for (int j = 0; j < 8; ++j) {
            int kk = kmap(j, h);
            pa[2*j+0] = ldsP[wave][ln][kk + 0];
            pa[2*j+1] = ldsP[wave][ln][kk + 1];
        }

        // O += P * V over the 32-key block, 4 d-tiles of 16
        #pragma unroll
        for (int nt = 0; nt < 4; ++nt) {
            v16bf vf = make_frag(Vt[nt * 16 + ln], h, 0);
            v8f acc = (nt == 0) ? o0 : (nt == 1) ? o1 : (nt == 2) ? o2 : o3;
            acc = __builtin_amdgcn_wmma_f32_16x16x32_bf16(false, pa, false, vf,
                                                          (short)0, acc, false, false);
            if (nt == 0) o0 = acc; else if (nt == 1) o1 = acc;
            else if (nt == 2) o2 = acc; else o3 = acc;
        }
    }

    // Normalize and store context rows
    float* cb = ctx + ((size_t)b * SEQ) * DVDIM;
    #pragma unroll
    for (int r = 0; r < 8; ++r) {
        float inv = 1.0f / lrow[r];
        float* cr = cb + (size_t)(qrow0 + r + 8 * h) * DVDIM;
        cr[ln +  0] = o0[r] * inv;
        cr[ln + 16] = o1[r] * inv;
        cr[ln + 32] = o2[r] * inv;
        cr[ln + 48] = o3[r] * inv;
    }
}

// ---------------------------------------------------------------------------
// Kernel 2: out[bt, m] = context[bt, :] . Wo[m, :] + b[m]
// [BT x 64] x [64 x 512]; each wave: 16 rows x 64 cols (4 tiles, 2 WMMA each)
// ---------------------------------------------------------------------------
__global__ __launch_bounds__(128) void out_proj_kernel(
    const float* __restrict__ ctx, const float* __restrict__ wo,
    const float* __restrict__ bias, float* __restrict__ out)
{
    const int tid  = threadIdx.x;
    const int wave = tid >> 5;
    const int lane = tid & 31;
    const int h    = lane >> 4;
    const int ln   = lane & 15;

    const int bt0 = blockIdx.x * 64 + wave * 16;
    const int m0  = blockIdx.y * 64;

    v16bf a0, a1;
    {
        const float* cr = ctx + (size_t)(bt0 + ln) * DVDIM;
        #pragma unroll
        for (int j = 0; j < 8; ++j) {
            int kk = kmap(j, h);
            a0[2*j+0] = (__bf16)cr[kk +  0];
            a0[2*j+1] = (__bf16)cr[kk +  1];
            a1[2*j+0] = (__bf16)cr[kk + 32];
            a1[2*j+1] = (__bf16)cr[kk + 33];
        }
    }

    #pragma unroll
    for (int nt = 0; nt < 4; ++nt) {
        int mcol = m0 + nt * 16 + ln;
        const float* wr = wo + (size_t)mcol * DVDIM;   // Wo row m, contiguous in v
        v16bf b0, b1;
        #pragma unroll
        for (int j = 0; j < 8; ++j) {
            int kk = kmap(j, h);
            b0[2*j+0] = (__bf16)wr[kk +  0];
            b0[2*j+1] = (__bf16)wr[kk +  1];
            b1[2*j+0] = (__bf16)wr[kk + 32];
            b1[2*j+1] = (__bf16)wr[kk + 33];
        }
        v8f acc = {};
        acc = __builtin_amdgcn_wmma_f32_16x16x32_bf16(false, a0, false, b0,
                                                      (short)0, acc, false, false);
        acc = __builtin_amdgcn_wmma_f32_16x16x32_bf16(false, a1, false, b1,
                                                      (short)0, acc, false, false);
        float bv = bias[mcol];
        #pragma unroll
        for (int r = 0; r < 8; ++r) {
            int row = bt0 + r + 8 * h;
            out[(size_t)row * DMDIM + mcol] = acc[r] + bv;
        }
    }
}

extern "C" void kernel_launch(void* const* d_in, const int* in_sizes, int n_in,
                              void* d_out, int out_size, void* d_ws, size_t ws_size,
                              hipStream_t stream) {
    const float* q  = (const float*)d_in[0];
    const float* k  = (const float*)d_in[1];
    const float* v  = (const float*)d_in[2];
    const float* wo = (const float*)d_in[3];
    const float* wb = (const float*)d_in[4];
    float* out = (float*)d_out;
    float* ctx = (float*)d_ws;   // B*T*DV f32 = 8.4 MB scratch

    dim3 g1(SEQ / 64, BATCH);
    flash_attn_kernel<<<g1, 128, 0, stream>>>(q, k, v, ctx);

    dim3 g2((BATCH * SEQ) / 64, DMDIM / 64);
    out_proj_kernel<<<g2, 128, 0, stream>>>(ctx, wo, wb, out);
}
